// SubgraphConv1D_55533927138052
// MI455X (gfx1250) — matrix operand order repaired and checked by reading, not compiled, not run on hardware
//
#include <hip/hip_runtime.h>

#define BSUB 1024
#define LPAD 64
#define SEGN 48
#define CIN  512
#define COUT 512
#define XS_STRIDE 520   // 512 + 8 f16 pad -> row stride = 260 dwords == 4 mod 64 banks
#define XS_ROWS   50    // row 0 = dense row -1 (zero), rows 1..48 = x rows 0..47, row 49 = zero

typedef __attribute__((ext_vector_type(16))) _Float16 v16h;
typedef __attribute__((ext_vector_type(8)))  _Float16 v8h;
typedef __attribute__((ext_vector_type(4)))  _Float16 v4h;
typedef __attribute__((ext_vector_type(8)))  float    v8f;
typedef __attribute__((ext_vector_type(4)))  float    v4f;

// --- K0: conv_w [COUT][CIN][3] f32  ->  wkn [3][COUT][CIN] f16 ------------
__global__ __launch_bounds__(256)
void prep_w_kernel(const float* __restrict__ w, _Float16* __restrict__ wkn) {
  int idx = blockIdx.x * 256 + threadIdx.x;       // over 3*512*512
  if (idx >= 3 * CIN * COUT) return;
  int ci = idx & (CIN - 1);
  int co = (idx >> 9) & (COUT - 1);
  int k  = idx >> 18;
  wkn[idx] = (_Float16)w[(co * CIN + ci) * 3 + k];
}

// --- K1: per-subgraph conv GEMM via WMMA; writes pre-BN f16 + partial stats
__global__ __launch_bounds__(256)
void conv_gemm_kernel(const float* __restrict__ x,
                      const _Float16* __restrict__ wkn,
                      _Float16* __restrict__ pre,
                      float* __restrict__ s1g,
                      float* __restrict__ s2g) {
  __shared__ __align__(16) _Float16 xs[XS_ROWS * XS_STRIDE];   // 52,000 B

  const int b   = blockIdx.x;
  const int tid = threadIdx.x;

  // Stage x[b*48 .. b*48+47][0..511] f32 -> f16 into LDS rows 1..48
  {
    const v4f* xg = (const v4f*)(x + (size_t)b * SEGN * CIN);  // 6144 float4
#pragma unroll
    for (int i = 0; i < 24; ++i) {
      int e   = tid + i * 256;
      int row = e >> 7;          // 128 float4 per row
      int c4  = e & 127;
      v4f g = xg[e];
      v4h h;
      h[0] = (_Float16)g[0]; h[1] = (_Float16)g[1];
      h[2] = (_Float16)g[2]; h[3] = (_Float16)g[3];
      *(v4h*)(&xs[(row + 1) * XS_STRIDE + c4 * 4]) = h;
    }
    // zero rows 0 and 49 (columns 0..519, 130 v4h each)
    if (tid < 130) {
      v4h z = {};
      *(v4h*)(&xs[tid * 4])                             = z;   // row 0
      *(v4h*)(&xs[(XS_ROWS - 1) * XS_STRIDE + tid * 4]) = z;   // row 49
    }
  }
  __syncthreads();

  const int wv    = tid >> 5;          // 8 waves
  const int lane  = tid & 31;
  const int rl    = lane & 15;         // A row-in-tile / C column-in-tile
  const int hi    = lane >> 4;         // half-wave select
  const int nwb   = wv * 64;           // this wave's channel base
  // mt=3 (output rows 48..63) is live only for k=0; lane rl=0 reads dense
  // row 47 (LDS row 48), all other lanes read the zero row 49 (broadcast).
  const int row3  = (rl == 0) ? (XS_ROWS - 2) : (XS_ROWS - 1);

  v8f acc[4][4] = {};                  // [m-tile][n-tile] 16x16 f32 frags

#pragma unroll
  for (int k = 0; k < 3; ++k) {
    const _Float16* wk = wkn + (size_t)k * CIN * COUT;
    for (int ci0 = 0; ci0 < CIN; ci0 += 32) {
      // A fragments (branch-free: zero rows supply the window padding)
      v16h a[4] = {};
#pragma unroll
      for (int mt = 0; mt < 4; ++mt) {
        if (mt < 3 || k == 0) {
          int rlds = (mt < 3) ? (16 * mt + rl + k) : row3;   // r_lds = r + 1
          const _Float16* p = &xs[rlds * XS_STRIDE + ci0 + hi * 8];
          v8h lo = *(const v8h*)p;
          v8h hh = *(const v8h*)(p + 16);
          a[mt] = __builtin_shufflevector(lo, hh, 0,1,2,3,4,5,6,7,
                                                   8,9,10,11,12,13,14,15);
        }
      }
      // B fragments: lane = channel column (rl), K in table order (hi*8, +16)
      v16h bf[4];
#pragma unroll
      for (int nt = 0; nt < 4; ++nt) {
        const _Float16* q = wk + (size_t)(nwb + nt * 16 + rl) * CIN + ci0 + hi * 8;
        v8h lo = *(const v8h*)q;
        v8h hh = *(const v8h*)(q + 16);
        bf[nt] = __builtin_shufflevector(lo, hh, 0,1,2,3,4,5,6,7,
                                                  8,9,10,11,12,13,14,15);
      }
#pragma unroll
      for (int nt = 0; nt < 4; ++nt) {
#pragma unroll
        for (int mt = 0; mt < 4; ++mt) {
          if (mt < 3 || k == 0) {
            acc[mt][nt] = __builtin_amdgcn_wmma_f32_16x16x32_f16(
                false, a[mt], false, bf[nt], (short)0, acc[mt][nt], false, false);
          }
        }
      }
    }
  }

  // Epilogue: store pre-BN f16 + deterministic per-(b,co) sum / sumsq
#pragma unroll
  for (int nt = 0; nt < 4; ++nt) {
    const int co = nwb + nt * 16 + rl;
    float ss = 0.f, sq = 0.f;
#pragma unroll
    for (int mt = 0; mt < 4; ++mt) {
#pragma unroll
      for (int r = 0; r < 8; ++r) {
        float v = acc[mt][nt][r];
        ss += v;
        sq += v * v;
        int l = 16 * mt + hi * 8 + r;
        pre[((size_t)b * LPAD + l) * COUT + co] = (_Float16)v;
      }
    }
    ss += __shfl_xor(ss, 16);
    sq += __shfl_xor(sq, 16);
    if (hi == 0) {
      s1g[b * COUT + co] = ss;
      s2g[b * COUT + co] = sq;
    }
  }
}

// --- K2: channel stats -> scale/shift (conv_b cancels under batch-stat BN)
__global__ __launch_bounds__(256)
void bn_stats_kernel(const float* __restrict__ s1g, const float* __restrict__ s2g,
                     const float* __restrict__ gamma, const float* __restrict__ beta,
                     float* __restrict__ scale, float* __restrict__ shift) {
  int co = blockIdx.x * 256 + threadIdx.x;
  if (co >= COUT) return;
  float s1 = 0.f, s2 = 0.f;
  for (int b = 0; b < BSUB; ++b) {
    s1 += s1g[b * COUT + co];
    s2 += s2g[b * COUT + co];
  }
  const float inv  = 1.0f / (float)(BSUB * LPAD);
  float mean = s1 * inv;
  float var  = s2 * inv - mean * mean;
  float sc   = gamma[co] * rsqrtf(var + 1e-5f);
  scale[co] = sc;
  shift[co] = beta[co] - mean * sc;
}

// --- K3: y = relu(pre*scale + shift) streamed to d_out ([(b*64+l), co]) ---
__global__ __launch_bounds__(256)
void bn_apply_kernel(const _Float16* __restrict__ pre,
                     const float* __restrict__ scale,
                     const float* __restrict__ shift,
                     float* __restrict__ out) {
  __shared__ float sS[COUT];
  __shared__ float sH[COUT];
  for (int i = threadIdx.x; i < COUT; i += 256) { sS[i] = scale[i]; sH[i] = shift[i]; }
  __syncthreads();
  size_t base = ((size_t)blockIdx.x * 256 + threadIdx.x) * 8;   // 8 elems/thread
  v8h p = *(const v8h*)(pre + base);
  int co0 = (int)(base & (COUT - 1));
#pragma unroll
  for (int j = 0; j < 8; ++j) {
    float v = (float)p[j] * sS[co0 + j] + sH[co0 + j];
    out[base + j] = v > 0.f ? v : 0.f;
  }
}

// --- K4: batch_out = idx // 64 (int32 bits) ------------------------------
__global__ __launch_bounds__(256)
void batch_out_kernel(int* __restrict__ bo) {
  int i = blockIdx.x * 256 + threadIdx.x;   // 65536 total
  bo[i] = i >> 6;
}

extern "C" void kernel_launch(void* const* d_in, const int* in_sizes, int n_in,
                              void* d_out, int out_size, void* d_ws, size_t ws_size,
                              hipStream_t stream) {
  const float* x      = (const float*)d_in[0];
  // d_in[1] = batch ids: layout is i//SEG by construction -> not needed
  const float* conv_w = (const float*)d_in[2];
  // d_in[3] = conv_b: cancels exactly under training-mode BatchNorm
  const float* gamma  = (const float*)d_in[4];
  const float* beta   = (const float*)d_in[5];

  char* ws = (char*)d_ws;
  const size_t wkn_off   = 0;                                   // 1.5 MB
  const size_t pre_off   = 2u << 20;                            // 67.1 MB
  const size_t pre_bytes = (size_t)BSUB * LPAD * COUT * sizeof(_Float16);
  const size_t s1_off    = pre_off + pre_bytes;                 // 2 MB
  const size_t s2_off    = s1_off + (size_t)BSUB * COUT * sizeof(float);
  const size_t sc_off    = s2_off + (size_t)BSUB * COUT * sizeof(float);
  const size_t sh_off    = sc_off + COUT * sizeof(float);

  _Float16* wkn = (_Float16*)(ws + wkn_off);
  _Float16* pre = (_Float16*)(ws + pre_off);
  float* s1g    = (float*)(ws + s1_off);
  float* s2g    = (float*)(ws + s2_off);
  float* scale  = (float*)(ws + sc_off);
  float* shift  = (float*)(ws + sh_off);

  float* out_x = (float*)d_out;
  int*   bo    = (int*)(out_x + (size_t)BSUB * LPAD * COUT);

  prep_w_kernel<<<(3 * CIN * COUT + 255) / 256, 256, 0, stream>>>(conv_w, wkn);
  conv_gemm_kernel<<<BSUB, 256, 0, stream>>>(x, wkn, pre, s1g, s2g);
  bn_stats_kernel<<<(COUT + 255) / 256, 256, 0, stream>>>(s1g, s2g, gamma, beta, scale, shift);
  bn_apply_kernel<<<(int)(((size_t)BSUB * LPAD * COUT) / (256 * 8)), 256, 0, stream>>>(pre, scale, shift, out_x);
  batch_out_kernel<<<(BSUB * LPAD + 255) / 256, 256, 0, stream>>>(bo);
}
